// GRAFFNet_14705968022044
// MI455X (gfx1250) — matrix operand order repaired and checked by reading, not compiled, not run on hardware
//
#include <hip/hip_runtime.h>
#include <hip/hip_bf16.h>
#include <math.h>

typedef __attribute__((ext_vector_type(16))) __bf16 v16bf;
typedef __attribute__((ext_vector_type(8)))  __bf16 v8bf;
typedef __attribute__((ext_vector_type(8)))  float  v8f;

#define NFEAT  128
#define NHID   64
#define NCLASS 40

// ---------------- utility kernels ----------------

__global__ void zero_f32(float* __restrict__ p, int n) {
    int i = blockIdx.x * blockDim.x + threadIdx.x;
    if (i < n) p[i] = 0.0f;
}

__global__ void init_deg(float* __restrict__ deg, int n) {
    int i = blockIdx.x * blockDim.x + threadIdx.x;
    if (i < n) deg[i] = 1.0f;   // self-loop contributes 1 to every node's degree
}

__global__ void deg_accum(const int* __restrict__ dst, int E, float* __restrict__ deg) {
    int e = blockIdx.x * blockDim.x + threadIdx.x;
    if (e < E) unsafeAtomicAdd(&deg[dst[e]], 1.0f);
}

__global__ void make_dinv(const float* __restrict__ deg, float* __restrict__ dinv, int n) {
    int i = blockIdx.x * blockDim.x + threadIdx.x;
    if (i < n) dinv[i] = rsqrtf(deg[i]);    // deg >= 1 always (self loops)
}

__global__ void f32_to_bf16(const float* __restrict__ in, __bf16* __restrict__ out, int n) {
    int i = blockIdx.x * blockDim.x + threadIdx.x;
    if (i < n) out[i] = (__bf16)in[i];
}

// Prepack a weight matrix W[K x Nact] (row-major, leading dim ldw) into the
// WMMA B-fragment-native layout: out[kstep][ntile][lane][e] (e contiguous),
// so each GEMM lane loads its 16 bf16 (32 B) with one vector load.
// B 32x16 fragment, contiguous k order used by this code (A loads match): lane = 16*(K/16) + N ; elem e -> K = k0 + (lane>>4)*16 + e.
// sym!=0: W is Q, use 0.5*(Q + Q^T). Cols >= Nact / rows >= K are zero-padded.
__global__ void pack_B(const float* __restrict__ W, int K, int Nact, int ldw,
                       int ksteps, int ntiles, int sym, __bf16* __restrict__ out) {
    int i = blockIdx.x * blockDim.x + threadIdx.x;
    int total = ksteps * ntiles * 32 * 16;
    if (i >= total) return;
    int e    = i & 15;
    int lane = (i >> 4) & 31;
    int nt   = (i >> 9) % ntiles;
    int s    = i / (512 * ntiles);
    int n = nt * 16 + (lane & 15);
    int k = s * 32 + (lane >> 4) * 16 + e;
    float val = 0.0f;
    if (n < Nact && k < K)
        val = sym ? 0.5f * (W[(size_t)k * ldw + n] + W[(size_t)n * ldw + k])
                  : W[(size_t)k * ldw + n];
    out[i] = (__bf16)val;
}

// ---------------- WMMA GEMM ----------------
// C[M x Nact] = A(bf16)[M x K] @ B (prepacked fragments). 256-thread blocks =
// 8 waves; each wave owns one 16-row M-tile and computes all NTILES 16-col
// tiles, reusing its A fragment. Fully unrolled -> KSTEPS*NTILES v_wmma ops.
template <int KSTEPS, int NTILES>
__global__ void __launch_bounds__(256)
wmma_gemm_bf16(const __bf16* __restrict__ A, int lda,
               const __bf16* __restrict__ Bpack,
               float* __restrict__ C, int ldc, int Nact,
               __bf16* __restrict__ Cbf, int mtiles) {
    const int lane = threadIdx.x & 31;
    const int wid  = threadIdx.x >> 5;
    const int mt   = blockIdx.x * 8 + wid;
    if (mt >= mtiles) return;            // wave-uniform: EXEC all-ones for WMMA
    const int half = lane >> 4;
    const int l16  = lane & 15;

    const __bf16* Arow = A + (size_t)(mt * 16 + l16) * lda;

    v8f acc[NTILES];
#pragma unroll
    for (int n = 0; n < NTILES; ++n) acc[n] = (v8f){};

#pragma unroll
    for (int s = 0; s < KSTEPS; ++s) {
        const int k0 = s * 32;
        // A 16x32 bf16 fragment: elems 0..15 = K[k0+half*16 ..] (matches packed B)
        union { v16bf v; v8bf h[2]; } af;
        af.h[0] = *(const v8bf*)(Arow + k0 + half * 16);
        af.h[1] = *(const v8bf*)(Arow + k0 + half * 16 + 8);
#pragma unroll
        for (int n = 0; n < NTILES; ++n) {
            const v16bf b =
                *(const v16bf*)(Bpack + ((size_t)(s * NTILES + n) * 32 + lane) * 16);
            acc[n] = __builtin_amdgcn_wmma_f32_16x16x32_bf16(
                false, af.v, false, b, (short)0, acc[n], false, false);
        }
    }

    // C/D layout: VGPR r, lanes 0-15 -> M=r, lanes 16-31 -> M=r+8; N = l16
#pragma unroll
    for (int n = 0; n < NTILES; ++n) {
        const int col = n * 16 + l16;
        if (col < Nact) {
#pragma unroll
            for (int r = 0; r < 8; ++r) {
                const int m = mt * 16 + r + half * 8;
                C[(size_t)m * ldc + col] = acc[n][r];
                if (Cbf) Cbf[(size_t)m * ldc + col] = (__bf16)acc[n][r];
            }
        }
    }
}

// ---------------- edge scatter: agg[dst] += hW[src] * dinv[src]*dinv[dst] ----------------
// Thread i: edge e = i/16, channel group g = i%16 (4 channels). Consecutive lanes ->
// consecutive channels of one edge: coalesced gather + coalesced hardware f32 atomics.
__global__ void edge_msg(const int* __restrict__ src, const int* __restrict__ dst,
                         const float* __restrict__ dinv, const float* __restrict__ hW,
                         float* __restrict__ agg, long long total) {
    long long i = (long long)blockIdx.x * blockDim.x + threadIdx.x;
    if (i >= total) return;
    const int g = (int)(i & 15);
    const long long e = i >> 4;
    const int s = src[e], d = dst[e];
    const float norm = dinv[s] * dinv[d];
    const float4 v = *(const float4*)(hW + (size_t)s * NHID + g * 4);
    float* out = agg + (size_t)d * NHID + g * 4;
    unsafeAtomicAdd(out + 0, v.x * norm);
    unsafeAtomicAdd(out + 1, v.y * norm);
    unsafeAtomicAdd(out + 2, v.z * norm);
    unsafeAtomicAdd(out + 3, v.w * norm);
}

// ---------------- GRAFF pointwise update ----------------
// hout = h + relu(agg + hW*dinv^2 (self-loop msg) - h*omega - beta*h0); also bf16 shadow.
__global__ void graff_update(const float* __restrict__ h, const float* __restrict__ h0,
                             const float* __restrict__ hW, const float* __restrict__ agg,
                             const float* __restrict__ dinv, const float* __restrict__ omega,
                             const float* __restrict__ beta_p, float* __restrict__ hout,
                             __bf16* __restrict__ hout_bf, int total) {
    int i = blockIdx.x * blockDim.x + threadIdx.x;
    if (i >= total) return;
    const int n = i >> 6;       // NHID == 64
    const int c = i & 63;
    const float di  = dinv[n];
    const float val = agg[i] + hW[i] * di * di - h[i] * omega[c] - beta_p[0] * h0[i];
    const float r   = h[i] + fmaxf(val, 0.0f);
    hout[i]    = r;
    hout_bf[i] = (__bf16)r;
}

// ---------------- log_softmax over 40 classes ----------------
__global__ void log_softmax40(const float* __restrict__ logits, float* __restrict__ out, int N) {
    int n = blockIdx.x * blockDim.x + threadIdx.x;
    if (n >= N) return;
    const float* row = logits + (size_t)n * NCLASS;
    float m = row[0];
#pragma unroll
    for (int j = 1; j < NCLASS; ++j) m = fmaxf(m, row[j]);
    float sum = 0.0f;
#pragma unroll
    for (int j = 0; j < NCLASS; ++j) sum += expf(row[j] - m);
    const float lse = m + logf(sum);
    float* orow = out + (size_t)n * NCLASS;
#pragma unroll
    for (int j = 0; j < NCLASS; ++j) orow[j] = row[j] - lse;
}

// ---------------- host orchestration ----------------

extern "C" void kernel_launch(void* const* d_in, const int* in_sizes, int n_in,
                              void* d_out, int out_size, void* d_ws, size_t ws_size,
                              hipStream_t stream) {
    const float* x     = (const float*)d_in[0];   // [N, 128]
    const int*   edge  = (const int*)d_in[1];     // [2, E]
    const float* Wenc  = (const float*)d_in[2];   // [128, 64]
    const float* omega = (const float*)d_in[3];   // [64]
    const float* Q     = (const float*)d_in[4];   // [64, 64]
    const float* beta  = (const float*)d_in[5];   // scalar
    const float* Wdec  = (const float*)d_in[6];   // [64, 40]

    const int N = in_sizes[0] / NFEAT;            // 100000
    const int E = in_sizes[1] / 2;                // 1600000
    const int* src = edge;
    const int* dst = edge + E;

    // workspace partition (256-byte aligned)
    char* base = (char*)d_ws;
    size_t off = 0;
    auto take = [&](size_t bytes) -> char* {
        char* p = base + off;
        off = (off + bytes + 255) & ~(size_t)255;
        return p;
    };
    float*  deg    = (float*)take((size_t)N * 4);
    float*  dinv   = (float*)take((size_t)N * 4);
    float*  h0     = (float*)take((size_t)N * NHID * 4);
    float*  hA     = (float*)take((size_t)N * NHID * 4);
    float*  hW     = (float*)take((size_t)N * NHID * 4);
    float*  agg    = (float*)take((size_t)N * NHID * 4);
    float*  logits = (float*)take((size_t)N * NCLASS * 4);
    __bf16* xbf    = (__bf16*)take((size_t)N * NFEAT * 2);
    __bf16* h0bf   = (__bf16*)take((size_t)N * NHID * 2);
    __bf16* hAbf   = (__bf16*)take((size_t)N * NHID * 2);
    __bf16* penc   = (__bf16*)take((size_t)4 * 4 * 512 * 2);  // KSTEPS=4, NTILES=4
    __bf16* psym   = (__bf16*)take((size_t)2 * 4 * 512 * 2);  // KSTEPS=2, NTILES=4
    __bf16* pdec   = (__bf16*)take((size_t)2 * 3 * 512 * 2);  // KSTEPS=2, NTILES=3
    (void)ws_size; (void)n_in; (void)out_size;

    const int B256 = 256;
    const int nh_total = N * NHID;
    const int mtiles = N / 16;                    // 6250
    const int gemm_blocks = (mtiles + 7) / 8;

    // degrees + normalization
    init_deg<<<(N + B256 - 1) / B256, B256, 0, stream>>>(deg, N);
    deg_accum<<<(E + B256 - 1) / B256, B256, 0, stream>>>(dst, E, deg);
    make_dinv<<<(N + B256 - 1) / B256, B256, 0, stream>>>(deg, dinv, N);

    // bf16 shadow of x; prepack weights into B-fragment layout
    f32_to_bf16<<<(N * NFEAT + B256 - 1) / B256, B256, 0, stream>>>(x, xbf, N * NFEAT);
    pack_B<<<(4 * 4 * 512 + B256 - 1) / B256, B256, 0, stream>>>(
        Wenc, NFEAT, NHID, NHID, 4, 4, 0, penc);
    pack_B<<<(2 * 4 * 512 + B256 - 1) / B256, B256, 0, stream>>>(
        Q, NHID, NHID, NHID, 2, 4, 1, psym);
    pack_B<<<(2 * 3 * 512 + B256 - 1) / B256, B256, 0, stream>>>(
        Wdec, NHID, NCLASS, NCLASS, 2, 3, 0, pdec);

    // encode: h0 = x @ W_enc  (K=128 -> 4 ksteps, N=64 -> 4 ntiles)
    wmma_gemm_bf16<4, 4><<<gemm_blocks, 256, 0, stream>>>(
        xbf, NFEAT, penc, h0, NHID, NHID, h0bf, mtiles);

    long long edge_total = (long long)E * 16;
    unsigned edge_blocks = (unsigned)((edge_total + B256 - 1) / B256);

    for (int step = 0; step < 2; ++step) {
        const float*  hcur   = (step == 0) ? h0 : hA;
        const __bf16* hcurbf = (step == 0) ? h0bf : hAbf;
        // hW = hcur @ W_sym  (K=64 -> 2 ksteps)
        wmma_gemm_bf16<2, 4><<<gemm_blocks, 256, 0, stream>>>(
            hcurbf, NHID, psym, hW, NHID, NHID, (__bf16*)nullptr, mtiles);
        // agg = scatter-add of normalized messages (non-self-loop edges)
        zero_f32<<<(nh_total + B256 - 1) / B256, B256, 0, stream>>>(agg, nh_total);
        edge_msg<<<edge_blocks, B256, 0, stream>>>(src, dst, dinv, hW, agg, edge_total);
        // h(out) = hcur + relu(agg + hW*dinv^2 - hcur*omega - beta*h0)
        graff_update<<<(nh_total + B256 - 1) / B256, B256, 0, stream>>>(
            hcur, h0, hW, agg, dinv, omega, beta, hA, hAbf, nh_total);
    }

    // decode: logits = hA @ W_dec  (K=64 -> 2 ksteps, N=40 padded to 48 -> 3 ntiles)
    wmma_gemm_bf16<2, 3><<<gemm_blocks, 256, 0, stream>>>(
        hAbf, NHID, pdec, logits, NCLASS, NCLASS, (__bf16*)nullptr, mtiles);

    log_softmax40<<<(N + B256 - 1) / B256, B256, 0, stream>>>(logits, (float*)d_out, N);
}
